// SimVectorQuantizer_72344429134123
// MI455X (gfx1250) — matrix operand order repaired and checked by reading, not compiled, not run on hardware
//
#include <hip/hip_runtime.h>
#include <hip/hip_bf16.h>
#include <math.h>

typedef __bf16 v16bf __attribute__((ext_vector_type(16)));
typedef __bf16 v8bf  __attribute__((ext_vector_type(8)));
typedef float  v8f   __attribute__((ext_vector_type(8)));
typedef float  v16f  __attribute__((ext_vector_type(16)));
typedef unsigned int u32x4 __attribute__((ext_vector_type(4)));
typedef int          i32x8 __attribute__((ext_vector_type(8)));
typedef int          i32x4 __attribute__((ext_vector_type(4)));

#define BQ 16
#define CQ 256
#define HQ 32
#define WQ 32
#define ROWS 16384   // B*H*W
#define NE 8192
#define DIM 256
#define TEMP_INV 100.0f   // 1/TEMP

#define NB_CHUNK 64                 // wn rows staged per TDM transfer
#define N_CHUNKS (NE / NB_CHUNK)    // 128
#define BROW 264                    // padded LDS row stride in bf16 elems (528B)

// ---------------------------------------------------------------------------
// TDM issue: 1-D 32KB chunk (4096 x 8B units) global->LDS, with LDS padding of
// 4 DWORDs every 128 DWORDs (one 512B row) to kill ds bank conflicts.
// D# per CDNA5 ISA. 6-arg builtin (clang-23 toolchain).
// ---------------------------------------------------------------------------
__device__ __forceinline__ void tdm_issue_chunk(const __bf16* gsrc, unsigned lds_off) {
  unsigned long long ga = (unsigned long long)(uintptr_t)gsrc;
  u32x4 g0;
  g0[0] = 1u;                                          // count=1, no gather
  g0[1] = lds_off;                                     // lds_addr
  g0[2] = (unsigned)ga;                                // global_addr[31:0]
  g0[3] = (unsigned)((ga >> 32) & 0x01FFFFFFu) | (2u << 30);  // addr[56:32] | type=2
  i32x8 g1;
  // data_size=8B(3), pad_enable=1, pad_interval=6 (128 DW), pad_amount=3 (4 DW)
  g1[0] = (3 << 16) | (1 << 20) | (6 << 22) | (3 << 25);
  g1[1] = 0x10000000;        // tensor_dim0[15:0]=0x1000 in bits[31:16]
  g1[2] = (1 << 16);         // tensor_dim0[31:16]=0 ; tensor_dim1=1 (lo16)
  g1[3] = 0x10000000;        // tensor_dim1 hi=0 ; tile_dim0=0x1000 in bits[31:16]
  g1[4] = 0;                 // tile_dim1=0 (unused), tile_dim2=0
  g1[5] = 4096;              // tensor_dim0_stride lo32
  g1[6] = 0;
  g1[7] = 0;
  i32x4 z4 = {0, 0, 0, 0};               // groups 2/3 unused (<=2D tensor)
  i32x8 z8 = {0, 0, 0, 0, 0, 0, 0, 0};   // trailing group (clang-23 form)
  __builtin_amdgcn_tensor_load_to_lds(g0, g1, z4, z4, z8, 0);
}

// ---------------------------------------------------------------------------
// Kernel 0: zero the global accumulators (must run every call; graph-safe)
// ---------------------------------------------------------------------------
__global__ void k_zero(float* __restrict__ g_avg, float* __restrict__ g_ent,
                       float* __restrict__ g_mse) {
  int i = blockIdx.x * 256 + threadIdx.x;
  if (i < NE) g_avg[i] = 0.f;
  if (i == 0) { g_ent[0] = 0.f; g_mse[0] = 0.f; }
}

// ---------------------------------------------------------------------------
// Kernel 1: transpose+normalize z: [B,C,H,W] -> zn rows [(b,h,w), c] f32+bf16
// ---------------------------------------------------------------------------
__global__ void k_znorm(const float* __restrict__ z,
                        float* __restrict__ zn_f32,
                        __bf16* __restrict__ zn_bf16) {
  __shared__ float tile[CQ * 33];
  __shared__ float red[256];
  __shared__ float inv[32];
  const int tid = threadIdx.x;
  const int bi  = blockIdx.x;          // b*32 + h
  const int b = bi >> 5, h = bi & 31;
  float ss = 0.f;
  const float* zb = z + ((size_t)b * CQ) * (HQ * WQ) + (size_t)h * WQ;
  for (int i = tid; i < CQ * 32; i += 256) {      // i%32 == tid%32 always
    int c = i >> 5, ww = i & 31;
    float v = zb[(size_t)c * (HQ * WQ) + ww];
    tile[c * 33 + ww] = v;
    ss += v * v;
  }
  red[tid] = ss;
  __syncthreads();
  if (tid < 32) {
    float s = 0.f;
    #pragma unroll
    for (int y = 0; y < 8; ++y) s += red[tid + 32 * y];
    inv[tid] = 1.0f / fmaxf(sqrtf(s), 1e-6f);
  }
  __syncthreads();
  const size_t rbase = (size_t)bi * 32;
  for (int i = tid; i < 32 * CQ; i += 256) {
    int ww = i >> 8, c = i & 255;
    float v = tile[c * 33 + ww] * inv[ww];
    size_t o = (rbase + ww) * DIM + c;
    zn_f32[o]  = v;
    zn_bf16[o] = (__bf16)v;
  }
}

// ---------------------------------------------------------------------------
// Kernel 2: weight = emb @ proj_w.T + proj_b  via bf16 WMMA, f32 accumulate
// ---------------------------------------------------------------------------
__global__ void __launch_bounds__(256) k_proj(const float* __restrict__ emb,
                                              const float* __restrict__ pw,
                                              const float* __restrict__ pb,
                                              float* __restrict__ weight) {
  const int tid = threadIdx.x;
  const int wave = tid >> 5, lane = tid & 31;
  const int hf = lane >> 4;
  const int lrow = lane & 15;
  const int k0 = blockIdx.x * 128 + wave * 16;

  v16bf a[8];
  {
    const float* ab = emb + (size_t)(k0 + lrow) * DIM;
    #pragma unroll
    for (int c = 0; c < 8; ++c) {
      const float* p = ab + c * 32 + hf * 16;
      v8f lo = *(const v8f*)(p);
      v8f hi = *(const v8f*)(p + 8);
      v16bf f;
      #pragma unroll
      for (int t = 0; t < 8; ++t) { f[t] = (__bf16)lo[t]; f[t + 8] = (__bf16)hi[t]; }
      a[c] = f;
    }
  }
  for (int d0 = 0; d0 < DIM; d0 += 16) {
    v8f acc = {};
    const float* bb = pw + (size_t)(d0 + lrow) * DIM + hf * 16;
    #pragma unroll
    for (int c = 0; c < 8; ++c) {
      v16f b32 = *(const v16f*)(bb + c * 32);
      v16bf bf;
      #pragma unroll
      for (int t = 0; t < 16; ++t) bf[t] = (__bf16)b32[t];
      acc = __builtin_amdgcn_wmma_f32_16x16x32_bf16(false, a[c], false, bf,
                                                    (short)0, acc, false, false);
    }
    float bias = pb[d0 + lrow];
    #pragma unroll
    for (int j = 0; j < 8; ++j)
      weight[(size_t)(k0 + j + 8 * hf) * DIM + d0 + lrow] = acc[j] + bias;
  }
}

// ---------------------------------------------------------------------------
// Kernel 3: normalize codebook rows in place; emit bf16 copy + w2 = ||wn||^2
// ---------------------------------------------------------------------------
__global__ void k_wnorm(float* __restrict__ wn, __bf16* __restrict__ wnb,
                        float* __restrict__ w2) {
  __shared__ float red[256];
  const int r = blockIdx.x, tid = threadIdx.x;
  float v = wn[(size_t)r * DIM + tid];
  red[tid] = v * v;
  __syncthreads();
  for (int s = 128; s > 0; s >>= 1) {
    if (tid < s) red[tid] += red[tid + s];
    __syncthreads();
  }
  float ss = red[0];
  float inv = 1.0f / fmaxf(sqrtf(ss), 1e-6f);
  float o = v * inv;
  wn[(size_t)r * DIM + tid]  = o;
  wnb[(size_t)r * DIM + tid] = (__bf16)o;
  if (tid == 0) w2[r] = ss * inv * inv;
}

// ---------------------------------------------------------------------------
// Kernel 4: fused GEMM + argmax + online softmax + avg_probs.
// 4 waves, 64 z-rows/block. wn streamed through LDS by TDM double-buffer.
// ---------------------------------------------------------------------------
__global__ void __launch_bounds__(128) k_main(
    const __bf16* __restrict__ znb, const __bf16* __restrict__ wnb,
    const float* __restrict__ w2, int* __restrict__ row_idx,
    float* __restrict__ g_avg, float* __restrict__ g_ent) {
  __shared__ __align__(16) __bf16 tileA[64 * DIM];        // 32 KB
  __shared__ float lds_avg[NE];                           // 32 KB
  __shared__ __align__(16) __bf16 tileB[2][NB_CHUNK * BROW]; // 2 x 33 KB, padded rows
  const int tid = threadIdx.x;
  const int wave = tid >> 5, lane = tid & 31;
  const int hf = lane >> 4, lrow = lane & 15;
  const int rowBase = blockIdx.x * 64;

  const unsigned ldsB0 = (unsigned)(uintptr_t)&tileB[0][0];
  const unsigned ldsB1 = (unsigned)(uintptr_t)&tileB[1][0];

  for (int i = tid; i < NE; i += 128) lds_avg[i] = 0.f;
  { // stage 64 zn rows (bf16) into LDS
    const uint4* src = (const uint4*)(znb + (size_t)rowBase * DIM);
    uint4* dst = (uint4*)tileA;
    for (int i = tid; i < (64 * DIM * 2) / 16; i += 128) dst[i] = src[i];
  }
  __syncthreads();

  // A fragments stay resident for BOTH passes
  v16bf a[8];
  {
    const __bf16* ab = tileA + (wave * 16 + lrow) * DIM;
    #pragma unroll
    for (int c = 0; c < 8; ++c) {
      const __bf16* p = ab + c * 32 + hf * 16;
      v8bf lo = *(const v8bf*)p;
      v8bf hi = *(const v8bf*)(p + 8);
      a[c] = __builtin_shufflevector(lo, hi, 0,1,2,3,4,5,6,7,8,9,10,11,12,13,14,15);
    }
  }

  float m[8], Z[8], S[8]; int idx[8];
  #pragma unroll
  for (int j = 0; j < 8; ++j) { m[j] = -INFINITY; Z[j] = 0.f; S[j] = 0.f; idx[j] = 0; }

  // ---- pass 1: dot + argmax + online softmax stats -------------------------
  if (wave == 0) tdm_issue_chunk(wnb, ldsB0);                       // chunk 0
  for (int ch = 0; ch < N_CHUNKS; ++ch) {
    if (wave == 0) {
      if (ch + 1 < N_CHUNKS) {
        tdm_issue_chunk(wnb + (size_t)(ch + 1) * NB_CHUNK * DIM,
                        (ch + 1) & 1 ? ldsB1 : ldsB0);
        __builtin_amdgcn_s_wait_tensorcnt(1);    // chunk ch complete
      } else {
        __builtin_amdgcn_s_wait_tensorcnt(0);
      }
    }
    __syncthreads();                             // LDS chunk visible to all
    const __bf16* btile = tileB[ch & 1];
    #pragma unroll
    for (int t = 0; t < 4; ++t) {
      const int n0 = ch * NB_CHUNK + t * 16;
      const __bf16* bb = btile + (t * 16 + lrow) * BROW + hf * 16;
      v8f acc = {};
      #pragma unroll
      for (int c = 0; c < 8; ++c) {
        v8bf lo = *(const v8bf*)(bb + c * 32);
        v8bf hi = *(const v8bf*)(bb + c * 32 + 8);
        v16bf bf = __builtin_shufflevector(lo, hi, 0,1,2,3,4,5,6,7,8,9,10,11,12,13,14,15);
        acc = __builtin_amdgcn_wmma_f32_16x16x32_bf16(false, a[c], false, bf,
                                                      (short)0, acc, false, false);
      }
      float w2s = TEMP_INV * w2[n0 + lrow];
      #pragma unroll
      for (int j = 0; j < 8; ++j) {
        float av = 2.0f * TEMP_INV * acc[j] - w2s;
        float mo = m[j];
        float mn = fmaxf(mo, av);
        float sc = __expf(mo - mn);
        float e  = __expf(av - mn);
        Z[j] = Z[j] * sc + e;
        S[j] = S[j] * sc + e * av;
        if (av > mo) idx[j] = n0 + lrow;
        m[j] = mn;
      }
    }
    __syncthreads();                             // buffer safe to overwrite
  }
  // ---- reduce partials across the 16 lanes sharing each row ----------------
  #pragma unroll
  for (int mask = 1; mask < 16; mask <<= 1) {
    #pragma unroll
    for (int j = 0; j < 8; ++j) {
      float m2 = __shfl_xor(m[j], mask, 32);
      float Z2 = __shfl_xor(Z[j], mask, 32);
      float S2 = __shfl_xor(S[j], mask, 32);
      int   i2 = __shfl_xor(idx[j], mask, 32);
      float mn = fmaxf(m[j], m2);
      float s1 = __expf(m[j] - mn), s2 = __expf(m2 - mn);
      Z[j] = Z[j] * s1 + Z2 * s2;
      S[j] = S[j] * s1 + S2 * s2;
      if (m2 > m[j]) idx[j] = i2;
      else if (m2 == m[j] && i2 < idx[j]) idx[j] = i2;  // argmin tie -> lowest k
      m[j] = mn;
    }
  }
  float Zi[8];
  #pragma unroll
  for (int j = 0; j < 8; ++j) Zi[j] = 1.0f / Z[j];
  if (lrow == 0) {
    float es = 0.f;
    #pragma unroll
    for (int j = 0; j < 8; ++j) {
      int r = rowBase + wave * 16 + j + 8 * hf;
      row_idx[r] = idx[j];
      es += m[j] + __logf(Z[j]) - S[j] * Zi[j];   // per-row sample entropy
    }
    atomicAdd(g_ent, es);
  }
  // ---- pass 2: recompute dot, scatter probs into avg accumulator -----------
  if (wave == 0) tdm_issue_chunk(wnb, ldsB0);
  for (int ch = 0; ch < N_CHUNKS; ++ch) {
    if (wave == 0) {
      if (ch + 1 < N_CHUNKS) {
        tdm_issue_chunk(wnb + (size_t)(ch + 1) * NB_CHUNK * DIM,
                        (ch + 1) & 1 ? ldsB1 : ldsB0);
        __builtin_amdgcn_s_wait_tensorcnt(1);
      } else {
        __builtin_amdgcn_s_wait_tensorcnt(0);
      }
    }
    __syncthreads();
    const __bf16* btile = tileB[ch & 1];
    #pragma unroll
    for (int t = 0; t < 4; ++t) {
      const int n0 = ch * NB_CHUNK + t * 16;
      const __bf16* bb = btile + (t * 16 + lrow) * BROW + hf * 16;
      v8f acc = {};
      #pragma unroll
      for (int c = 0; c < 8; ++c) {
        v8bf lo = *(const v8bf*)(bb + c * 32);
        v8bf hi = *(const v8bf*)(bb + c * 32 + 8);
        v16bf bf = __builtin_shufflevector(lo, hi, 0,1,2,3,4,5,6,7,8,9,10,11,12,13,14,15);
        acc = __builtin_amdgcn_wmma_f32_16x16x32_bf16(false, a[c], false, bf,
                                                      (short)0, acc, false, false);
      }
      float w2s = TEMP_INV * w2[n0 + lrow];
      float sl = 0.f;
      #pragma unroll
      for (int j = 0; j < 8; ++j) {
        float av = 2.0f * TEMP_INV * acc[j] - w2s;
        sl += __expf(av - m[j]) * Zi[j];
      }
      sl += __shfl_xor(sl, 16, 32);              // fold both half-wave rows
      if (hf == 0) atomicAdd(&lds_avg[n0 + lrow], sl);
    }
    __syncthreads();
  }
  for (int i = tid; i < NE; i += 128) {
    float v = lds_avg[i];
    if (v != 0.f) atomicAdd(&g_avg[i], v);
  }
}

// ---------------------------------------------------------------------------
// Kernel 5: z_q output (gather + NHWC->NCHW) + MSE accumulation
// ---------------------------------------------------------------------------
__global__ void k_output(const float* __restrict__ wn,
                         const float* __restrict__ w2,
                         const float* __restrict__ zn,
                         const int* __restrict__ row_idx,
                         float* __restrict__ out_zq,
                         float* __restrict__ g_mse) {
  const int tid = threadIdx.x;
  const int bi = blockIdx.x;                 // b*32 + h
  const int b = bi >> 5, h = bi & 31;
  const int w = tid & 31, cg = tid >> 5;
  const int r = bi * 32 + w;
  const int id = row_idx[r];
  const float qinv = 1.0f / fmaxf(sqrtf(w2[id]), 1e-6f);
  const float* wrow = wn + (size_t)id * DIM;
  const float* zrow = zn + (size_t)r * DIM;
  float ms = 0.f;
  for (int i = 0; i < 32; ++i) {
    int c = cg * 32 + i;
    float qv = wrow[c];
    out_zq[(((size_t)b * CQ + c) * HQ + h) * WQ + w] = qv;
    float d = qv * qinv - zrow[c];
    ms += d * d;
  }
  #pragma unroll
  for (int mask = 1; mask < 32; mask <<= 1) ms += __shfl_xor(ms, mask, 32);
  if ((tid & 31) == 0) atomicAdd(g_mse, ms);
}

// ---------------------------------------------------------------------------
// Kernel 6: final scalar losses
// ---------------------------------------------------------------------------
__global__ void k_final(const float* __restrict__ g_avg,
                        const float* __restrict__ g_ent,
                        const float* __restrict__ g_mse,
                        float* __restrict__ out_sc) {
  __shared__ float red[256];
  const int tid = threadIdx.x;
  float s = 0.f;
  for (int i = tid; i < NE; i += 256) {
    float p = g_avg[i] * (1.0f / (float)ROWS);
    s += p * __logf(p + 1e-5f);
  }
  red[tid] = s;
  __syncthreads();
  for (int t = 128; t > 0; t >>= 1) { if (tid < t) red[tid] += red[tid + t]; __syncthreads(); }
  if (tid == 0) {
    float avg_entropy = -red[0];
    float sample_entropy = g_ent[0] * (1.0f / (float)ROWS);
    float mse = g_mse[0] * (1.0f / ((float)ROWS * (float)DIM));
    out_sc[0] = mse;                          // codebook_loss
    out_sc[1] = 0.25f * mse;                  // commit_loss
    out_sc[2] = 0.1f * (sample_entropy - avg_entropy);  // entropy_loss
  }
}

// ---------------------------------------------------------------------------
// Kernel 7: flat_indices as float into d_out tail
// ---------------------------------------------------------------------------
__global__ void k_idx(const int* __restrict__ row_idx, float* __restrict__ out) {
  int i = blockIdx.x * 256 + threadIdx.x;
  if (i < ROWS) out[i] = (float)row_idx[i];
}

// ---------------------------------------------------------------------------
extern "C" void kernel_launch(void* const* d_in, const int* in_sizes, int n_in,
                              void* d_out, int out_size, void* d_ws, size_t ws_size,
                              hipStream_t stream) {
  const float* z   = (const float*)d_in[0];
  const float* emb = (const float*)d_in[1];
  const float* pw  = (const float*)d_in[2];
  const float* pb  = (const float*)d_in[3];

  char* ws = (char*)d_ws;
  size_t off = 0;
  auto alloc = [&](size_t bytes) -> void* {
    void* p = ws + off;
    off = (off + bytes + 255) & ~(size_t)255;
    return p;
  };
  float*  zn_f32  = (float*) alloc((size_t)ROWS * DIM * 4);
  __bf16* zn_bf16 = (__bf16*)alloc((size_t)ROWS * DIM * 2);
  float*  wn_f32  = (float*) alloc((size_t)NE * DIM * 4);
  __bf16* wn_bf16 = (__bf16*)alloc((size_t)NE * DIM * 2);
  float*  w2      = (float*) alloc((size_t)NE * 4);
  int*    row_idx = (int*)   alloc((size_t)ROWS * 4);
  float*  g_avg   = (float*) alloc((size_t)NE * 4);
  float*  g_ent   = (float*) alloc(256);
  float*  g_mse   = (float*) alloc(256);

  float* out_zq  = (float*)d_out;
  float* out_sc  = out_zq + (size_t)BQ * CQ * HQ * WQ;
  float* out_idx = out_sc + 3;

  hipLaunchKernelGGL(k_zero,   dim3(NE / 256), dim3(256), 0, stream, g_avg, g_ent, g_mse);
  hipLaunchKernelGGL(k_znorm,  dim3(BQ * HQ),  dim3(256), 0, stream, z, zn_f32, zn_bf16);
  hipLaunchKernelGGL(k_proj,   dim3(NE / 128), dim3(256), 0, stream, emb, pw, pb, wn_f32);
  hipLaunchKernelGGL(k_wnorm,  dim3(NE),       dim3(256), 0, stream, wn_f32, wn_bf16, w2);
  hipLaunchKernelGGL(k_main,   dim3(ROWS / 64), dim3(128), 0, stream,
                     zn_bf16, wn_bf16, w2, row_idx, g_avg, g_ent);
  hipLaunchKernelGGL(k_output, dim3(BQ * HQ),  dim3(256), 0, stream,
                     wn_f32, w2, zn_f32, row_idx, out_zq, g_mse);
  hipLaunchKernelGGL(k_final,  dim3(1),        dim3(256), 0, stream, g_avg, g_ent, g_mse, out_sc);
  hipLaunchKernelGGL(k_idx,    dim3(ROWS / 256), dim3(256), 0, stream, row_idx, out_idx);
}
